// InterMemory_79276506349970
// MI455X (gfx1250) — compile-verified, matrix operand order checked by reading, not run on hardware
//
#include <hip/hip_runtime.h>
#include <hip/hip_bf16.h>

typedef __bf16 v16bf __attribute__((ext_vector_type(16)));
typedef float  v8f   __attribute__((ext_vector_type(8)));
typedef float  v4f   __attribute__((ext_vector_type(4)));
typedef unsigned int v4u __attribute__((ext_vector_type(4)));

#define B_   32
#define L_   4096
#define D_   512
#define M_   64
#define H_   8
#define HD_  64
#define ROWS_PER_BLK 64

__device__ __forceinline__ unsigned short f2bf(float f) {
  unsigned u = __float_as_uint(f);
  u += 0x7FFFu + ((u >> 16) & 1u);      // round-to-nearest-even
  return (unsigned short)(u >> 16);
}
__device__ __forceinline__ unsigned pack2bf(float lo, float hi) {
  return (unsigned)f2bf(lo) | ((unsigned)f2bf(hi) << 16);
}

// 16-bit A/B fragment K mapping (ISA 7.12.2): lane<16 -> K {0..7,16..23},
// lane>=16 -> K {8..15,24..31}; j = element index within the 16 bf16 of v16bf.
__device__ __forceinline__ int klocal_of(int lane, int j) {
  return (j & 7) + ((lane >> 4) << 3) + ((j >> 3) << 4);
}
__device__ __forceinline__ void inv_klocal(int kl, int& half, int& j) {
  half = (kl >> 3) & 1;
  j = (kl & 7) | ((kl >> 4) << 3);
}

// ---------------------------------------------------------------------------
// Prep 1: convert Wq / Wo (fp32 row-major [out,in]) into bf16 fragments laid
// out exactly as the WMMA B operand wants them (frag = nt*16+kt, 512 bf16
// per frag, lane-major: lane*16 + j). B[k,n] = W^T[k,n] = W[n,k].
// ---------------------------------------------------------------------------
__global__ void prep_w_kernel(const float* __restrict__ Wq,
                              const float* __restrict__ Wo,
                              unsigned short* __restrict__ wq_swz,
                              unsigned short* __restrict__ wo_swz) {
  int idx = blockIdx.x * blockDim.x + threadIdx.x;          // 0 .. 2*262144-1
  const float* W = (idx < 262144) ? Wq : Wo;
  unsigned short* dst = (idx < 262144) ? wq_swz : wo_swz;
  int e = idx & 262143;
  int frag = e >> 9;               // nt*16 + kt
  int within = e & 511;
  int lane = within >> 4;
  int j = within & 15;
  int nt = frag >> 4, kt = frag & 15;
  int n = nt * 16 + (lane & 15);
  int k = kt * 32 + klocal_of(lane, j);
  dst[e] = f2bf(W[n * D_ + k]);
}

// ---------------------------------------------------------------------------
// Prep 2: K = memory@Wk^T+bk, V = memory@Wv^T+bv, stored directly as per-head
// B fragments.
//   Kswz (for S = Q_h @ K_h^T): B[k=feat, n=mem slot] = Kproj[n, h*64+k]
//   Vswz (for ctx = P @ V_h)  : B[k=mem slot, n=feat] = Vproj[k, h*64+n]
// frag index within head = nt*2 + kt2 (4 ntiles x 2 ktiles), head stride 8.
// ---------------------------------------------------------------------------
__global__ void prep_kv_kernel(const float* __restrict__ memory,
                               const float* __restrict__ Wk, const float* __restrict__ bk,
                               const float* __restrict__ Wv, const float* __restrict__ bv,
                               unsigned short* __restrict__ k_swz,
                               unsigned short* __restrict__ v_swz) {
  int idx = blockIdx.x * blockDim.x + threadIdx.x;          // 0..65535
  bool isV = idx >= 32768;
  int e = idx & 32767;
  int h = e >> 12;
  int rem = e & 4095;
  int frag = rem >> 9;             // nt*2 + kt2
  int within = rem & 511;
  int lane = within >> 4, j = within & 15;
  int nt = frag >> 1, kt2 = frag & 1;
  int nloc = nt * 16 + (lane & 15);
  int kloc = kt2 * 32 + klocal_of(lane, j);
  float acc;
  if (!isV) {
    int feat = h * HD_ + kloc;     // K: n = mem slot, k = head feature
    const float* mrow = memory + nloc * D_;
    const float* wrow = Wk + feat * D_;
    acc = bk[feat];
    for (int d = 0; d < D_; ++d) acc = fmaf(mrow[d], wrow[d], acc);
    k_swz[e] = f2bf(acc);
  } else {
    int feat = h * HD_ + nloc;     // V: n = head feature, k = mem slot
    const float* mrow = memory + kloc * D_;
    const float* wrow = Wv + feat * D_;
    acc = bv[feat];
    for (int d = 0; d < D_; ++d) acc = fmaf(mrow[d], wrow[d], acc);
    v_swz[e] = f2bf(acc);
  }
}

// ---------------------------------------------------------------------------
// Fused: Q-proj -> per-head attention (softmax over 64 keys) -> ctx -> O-proj
// One block = one batch element x 64 query rows. 256 threads = 8 wave32.
// ---------------------------------------------------------------------------
__global__ __launch_bounds__(256) void attn_kernel(
    const float* __restrict__ X, const int* __restrict__ maskp,
    const float* __restrict__ bq, const float* __restrict__ bo,
    const unsigned short* __restrict__ wq_swz,
    const unsigned short* __restrict__ wo_swz,
    const unsigned short* __restrict__ k_swz,
    const unsigned short* __restrict__ v_swz,
    float* __restrict__ out) {
  extern __shared__ unsigned short lds16[];
  // LDS map (ushort units): X frags 32K, Q frags 32K, ctx frags 32K, P 8K
  const int BUFX = 0;
  const int BUFQ = 32768;
  const int BUFC = 65536;
  const int BUFP = 98304;

  const int tid = threadIdx.x;
  const int lane = tid & 31;
  const int w = tid >> 5;
  const int hl = lane >> 4;
  const int bi = blockIdx.y;
  const int rowbase = blockIdx.x * ROWS_PER_BLK;

  // ---- Stage A: X tile (64x512 fp32) -> bf16 A-fragments in LDS.
  // Exploit layout contiguity: for a fixed (lane, j-half) the 8 K values are
  // contiguous in memory, so each group = 2x b128 global loads + 1x b128 LDS
  // store of packed bf16.
  for (int g = tid; g < 4096; g += 256) {
    int frag = g >> 6;                   // mt*16 + kt  (64 groups per frag)
    int gin = g & 63;
    int wl = gin >> 1;                   // fragment lane 0..31
    int jh = gin & 1;                    // j-half: j in [0,8) or [8,16)
    int mt = frag >> 4, kt = frag & 15;
    int m = mt * 16 + (wl & 15);
    int kbase = kt * 32 + ((wl >> 4) << 3) + (jh << 4);   // klocal_of(wl, jh*8)
    const float* src = &X[(size_t)(bi * L_ + rowbase + m) * D_ + kbase];
    v4f x0 = *reinterpret_cast<const v4f*>(src);
    v4f x1 = *reinterpret_cast<const v4f*>(src + 4);
    v4u p;
    p.x = pack2bf(x0.x, x0.y);
    p.y = pack2bf(x0.z, x0.w);
    p.z = pack2bf(x1.x, x1.y);
    p.w = pack2bf(x1.z, x1.w);
    *reinterpret_cast<v4u*>(&lds16[BUFX + frag * 512 + wl * 16 + jh * 8]) = p;
  }
  __syncthreads();

  // ---- Stage B: Q = X @ Wq^T + bq ; wave w owns rows mt=w>>1, 16 col-tiles
  {
    const int mt = w >> 1;
    const int ntb = (w & 1) * 16;
    for (int nti = 0; nti < 16; ++nti) {
      const int nt = ntb + nti;
      if (nti + 1 < 16)
        __builtin_prefetch(&wq_swz[(size_t)((ntb + nti + 1) * 16) * 512 + lane * 16], 0, 0);
      v8f c = {};
      for (int kt = 0; kt < 16; ++kt) {
        v16bf a = *reinterpret_cast<const v16bf*>(&lds16[BUFX + (mt * 16 + kt) * 512 + lane * 16]);
        v16bf bm = *reinterpret_cast<const v16bf*>(&wq_swz[(size_t)(nt * 16 + kt) * 512 + lane * 16]);
        c = __builtin_amdgcn_wmma_f32_16x16x32_bf16(false, a, false, bm, (short)0, c, false, false);
      }
      const int n = nt * 16 + (lane & 15);
      const float bias = bq[n];
      int half, j2; inv_klocal(n & 31, half, j2);
      const int fragbase = BUFQ + (mt * 16 + (n >> 5)) * 512;
      #pragma unroll
      for (int g = 0; g < 8; ++g) {
        int r = g + 8 * hl;              // row within 16
        lds16[fragbase + (r + 16 * half) * 16 + j2] = f2bf(c[g] + bias);
      }
    }
  }
  __syncthreads();

  // ---- Stage C/D: head h = wave id; 16-row blocks sequentially
  {
    const int h = w;
    for (int mt = 0; mt < 4; ++mt) {
      v8f s[4];
      for (int nt = 0; nt < 4; ++nt) {
        v8f c = {};
        for (int k2 = 0; k2 < 2; ++k2) {
          v16bf a = *reinterpret_cast<const v16bf*>(&lds16[BUFQ + (mt * 16 + 2 * h + k2) * 512 + lane * 16]);
          v16bf bm = *reinterpret_cast<const v16bf*>(&k_swz[(size_t)(h * 8 + nt * 2 + k2) * 512 + lane * 16]);
          c = __builtin_amdgcn_wmma_f32_16x16x32_bf16(false, a, false, bm, (short)0, c, false, false);
        }
        s[nt] = c;
      }
      // mask bias + softmax over 64 keys (4 nt accums x 16 lanes-in-half)
      float pbias[8];
      #pragma unroll
      for (int g = 0; g < 8; ++g) {
        int row = rowbase + mt * 16 + g + 8 * hl;
        pbias[g] = maskp[bi * L_ + row] ? 0.0f : -1e9f;
      }
      const float scale = 0.125f;        // 1/sqrt(64)
      #pragma unroll
      for (int g = 0; g < 8; ++g) {
        float mx = -3.4e38f;
        #pragma unroll
        for (int nt = 0; nt < 4; ++nt) {
          s[nt][g] = s[nt][g] * scale + pbias[g];
          mx = fmaxf(mx, s[nt][g]);
        }
        mx = fmaxf(mx, __shfl_xor(mx, 1, 32));
        mx = fmaxf(mx, __shfl_xor(mx, 2, 32));
        mx = fmaxf(mx, __shfl_xor(mx, 4, 32));
        mx = fmaxf(mx, __shfl_xor(mx, 8, 32));
        float sum = 0.0f;
        #pragma unroll
        for (int nt = 0; nt < 4; ++nt) {
          s[nt][g] = __expf(s[nt][g] - mx);
          sum += s[nt][g];
        }
        sum += __shfl_xor(sum, 1, 32);
        sum += __shfl_xor(sum, 2, 32);
        sum += __shfl_xor(sum, 4, 32);
        sum += __shfl_xor(sum, 8, 32);
        float inv = 1.0f / sum;
        #pragma unroll
        for (int nt = 0; nt < 4; ++nt) s[nt][g] *= inv;
      }
      // P -> bf16 A-fragments, per-wave region (wave-local, DS is in-order)
      #pragma unroll
      for (int nt = 0; nt < 4; ++nt) {
        int slot = nt * 16 + (lane & 15);
        int half, j2; inv_klocal(slot & 31, half, j2);
        int fb = BUFP + w * 1024 + (slot >> 5) * 512;
        #pragma unroll
        for (int g = 0; g < 8; ++g) {
          int r = g + 8 * hl;
          lds16[fb + (r + 16 * half) * 16 + j2] = f2bf(s[nt][g]);
        }
      }
      // ctx = P @ V_h -> bf16 ctx fragments
      for (int nt = 0; nt < 4; ++nt) {
        v8f c = {};
        for (int k2 = 0; k2 < 2; ++k2) {
          v16bf a = *reinterpret_cast<const v16bf*>(&lds16[BUFP + w * 1024 + k2 * 512 + lane * 16]);
          v16bf bm = *reinterpret_cast<const v16bf*>(&v_swz[(size_t)(h * 8 + nt * 2 + k2) * 512 + lane * 16]);
          c = __builtin_amdgcn_wmma_f32_16x16x32_bf16(false, a, false, bm, (short)0, c, false, false);
        }
        int n = h * 64 + nt * 16 + (lane & 15);
        int half, j2; inv_klocal(n & 31, half, j2);
        int fb = BUFC + (mt * 16 + (n >> 5)) * 512;
        #pragma unroll
        for (int g = 0; g < 8; ++g) {
          int r = g + 8 * hl;
          lds16[fb + (r + 16 * half) * 16 + j2] = f2bf(c[g]);
        }
      }
    }
  }
  __syncthreads();

  // ---- Stage E: out = ctx @ Wo^T + bo (fp32 stores)
  {
    const int mt = w >> 1;
    const int ntb = (w & 1) * 16;
    for (int nti = 0; nti < 16; ++nti) {
      const int nt = ntb + nti;
      if (nti + 1 < 16)
        __builtin_prefetch(&wo_swz[(size_t)((ntb + nti + 1) * 16) * 512 + lane * 16], 0, 0);
      v8f c = {};
      for (int kt = 0; kt < 16; ++kt) {
        v16bf a = *reinterpret_cast<const v16bf*>(&lds16[BUFC + (mt * 16 + kt) * 512 + lane * 16]);
        v16bf bm = *reinterpret_cast<const v16bf*>(&wo_swz[(size_t)(nt * 16 + kt) * 512 + lane * 16]);
        c = __builtin_amdgcn_wmma_f32_16x16x32_bf16(false, a, false, bm, (short)0, c, false, false);
      }
      const int n = nt * 16 + (lane & 15);
      const float bias = bo[n];
      #pragma unroll
      for (int g = 0; g < 8; ++g) {
        int m = mt * 16 + g + 8 * hl;
        out[(size_t)(bi * L_ + rowbase + m) * D_ + n] = c[g] + bias;
      }
    }
  }
}

// ---------------------------------------------------------------------------
extern "C" void kernel_launch(void* const* d_in, const int* in_sizes, int n_in,
                              void* d_out, int out_size, void* d_ws, size_t ws_size,
                              hipStream_t stream) {
  const float* X   = (const float*)d_in[0];
  const int*   msk = (const int*)d_in[1];
  const float* mem = (const float*)d_in[2];
  const float* Wq  = (const float*)d_in[3];
  const float* bq  = (const float*)d_in[4];
  const float* Wk  = (const float*)d_in[5];
  const float* bk  = (const float*)d_in[6];
  const float* Wv  = (const float*)d_in[7];
  const float* bv  = (const float*)d_in[8];
  const float* Wo  = (const float*)d_in[9];
  const float* bo  = (const float*)d_in[10];
  float* out = (float*)d_out;

  unsigned short* wq_swz = (unsigned short*)d_ws;
  unsigned short* wo_swz = wq_swz + 512 * 512;
  unsigned short* k_swz  = wo_swz + 512 * 512;
  unsigned short* v_swz  = k_swz + 8 * 8 * 512;

  prep_w_kernel<<<2048, 256, 0, stream>>>(Wq, Wo, wq_swz, wo_swz);
  prep_kv_kernel<<<256, 256, 0, stream>>>(mem, Wk, bk, Wv, bv, k_swz, v_swz);

  const size_t shmem = 212992;   // 208 KB dynamic LDS (WGP allows 320 KB)
  hipFuncSetAttribute(reinterpret_cast<const void*>(attn_kernel),
                      hipFuncAttributeMaxDynamicSharedMemorySize, (int)shmem);
  attn_kernel<<<dim3(L_ / ROWS_PER_BLK, B_), 256, shmem, stream>>>(
      X, msk, bq, bo, wq_swz, wo_swz, k_swz, v_swz, out);
}
